// Navigator_44083544326733
// MI455X (gfx1250) — matrix operand order repaired and checked by reading, not compiled, not run on hardware
//
#include <hip/hip_runtime.h>
#include <hip/hip_bf16.h>

// ---- problem constants -----------------------------------------------------
#define PB 8
#define PS 512
#define PC 32
#define PH 256
#define PM (PB*PS*PC)        // 131072 rows of cand
#define PBS (PB*PS)          // 4096 (b,s) rows
#define MT 128               // rows per workgroup tile (main kernel)
#define NTILES_K 16          // k_proj N tiles (256 cols)
#define NTILES_T 4           // t2 N tiles (64 cols)
#define NTILES_ALL 20
#define PANEL_K 64           // K-panel staged in LDS (bf16 elems)
#define LDSW_LD 80           // 64 bf16 + 16 pad per LDS W row (160 B pitch)

typedef __attribute__((ext_vector_type(16))) __bf16 bf16x16;
typedef __attribute__((ext_vector_type(8)))  float  floatx8;
typedef __attribute__((ext_vector_type(4)))  unsigned int u32x4;
typedef __attribute__((ext_vector_type(8)))  int i32x8;
typedef __attribute__((ext_vector_type(4)))  int i32x4;

struct U32B { uint4 a, b; };   // 32-byte chunk for LDS bf16 fragment loads

#define WMMA_BF16(A,Bf,Cc) \
  __builtin_amdgcn_wmma_f32_16x16x32_bf16(false,(A),false,(Bf),(short)0,(Cc),false,false)

#define HAS_TDM __has_builtin(__builtin_amdgcn_tensor_load_to_lds)

__device__ __forceinline__ float geluf(float x) {
  return 0.5f * x * (1.0f + erff(x * 0.70710678118654752f));
}
__device__ __forceinline__ float sigmoidf_(float x) {
  return 1.0f / (1.0f + __expf(-x));
}

// A fragment (16x32 bf16): lane&15 = row (rowk points at that row + kbase),
// lane>>4 selects K sub-chunks {0..15} vs {16..31}.
__device__ __forceinline__ bf16x16 load_a_frag_f32(const float* rowk, int half) {
  bf16x16 a;
  const float* p0 = rowk + half * 16;
#pragma unroll
  for (int i = 0; i < 16; ++i) a[i] = (__bf16)p0[i];
  return a;
}
// B fragment (32x16 bf16) from f32 global: p points at W[row n][kbase + half*16].
__device__ __forceinline__ bf16x16 load_b_frag_f32(const float* p) {
  bf16x16 b;
#pragma unroll
  for (int i = 0; i < 16; ++i) b[i] = (__bf16)p[i];
  return b;
}
// B fragment from bf16 LDS (16-byte aligned, 32 contiguous bytes).
__device__ __forceinline__ bf16x16 load_b_frag_lds(const __bf16* p) {
  union { U32B s; bf16x16 v; } cv;
  cv.s = *(const U32B*)p;
  return cv.v;
}
__device__ __forceinline__ floatx8 zero8() {
  floatx8 z;
#pragma unroll
  for (int i = 0; i < 8; ++i) z[i] = 0.0f;
  return z;
}

// ---- kernel P: fold dp/ap branches of Wk into rank-1 vectors ---------------
__global__ __launch_bounds__(256) void nav_pre(const float* __restrict__ Wk,
                                               const float* __restrict__ Wd,
                                               const float* __restrict__ bd,
                                               const float* __restrict__ Wa,
                                               const float* __restrict__ ba,
                                               float* vecd, float* veca, float* kcst) {
  int n = threadIdx.x;            // 0..255
  const float* row = Wk + n * (3 * PH);
  float vd = 0.f, va = 0.f, kc = 0.f;
  for (int k = 0; k < PH; ++k) {
    float w1 = row[PH + k];       // Wk[n, 256+k]  (dp block)
    float w2 = row[2 * PH + k];   // Wk[n, 512+k]  (ap block)
    vd += w1 * Wd[k];
    va += w2 * Wa[k];
    kc += w1 * bd[k] + w2 * ba[k];
  }
  vecd[n] = vd; veca[n] = va; kcst[n] = kc;
}

// ---- kernel W: one-time f32 -> bf16 conversion of the fused weight matrix --
// Wbf is row-major [320][256]: rows 0..255 = Wk[:, :256], rows 256..319 = Wt2.
__global__ __launch_bounds__(256) void nav_cvtw(const float* __restrict__ Wk,
                                                const float* __restrict__ Wt2,
                                                __bf16* __restrict__ Wbf) {
  for (int e = blockIdx.x * 256 + threadIdx.x; e < 320 * 256; e += gridDim.x * 256) {
    int n = e >> 8, k = e & 255;
    float v = (n < PH) ? Wk[(size_t)n * (3 * PH) + k]
                       : Wt2[(size_t)(n - PH) * PH + k];
    Wbf[e] = (__bf16)v;
  }
}

// ---- kernel A: per-(b,s) GEMMs: q_proj, qr, t1g ----------------------------
__global__ __launch_bounds__(256) void nav_bs(const float* __restrict__ traj,
                                              const float* __restrict__ dest,
                                              const float* __restrict__ Wq,
                                              const float* __restrict__ Wr1,
                                              const float* __restrict__ Wt1,
                                              const float* __restrict__ bt1,
                                              const float* __restrict__ Wt3,
                                              float* __restrict__ qproj,
                                              float* __restrict__ qrw,
                                              float* __restrict__ t1g) {
  const int wave = threadIdx.x >> 5;
  const int lane = threadIdx.x & 31;
  const int half = lane >> 4;
  const int ln   = lane & 15;
  const int wid  = blockIdx.x * 8 + wave;     // 0..255
  const int m0   = wid * 16;                  // 16 (b,s) rows per wave
  const int mrow = m0 + ln;
  const int b    = mrow >> 9;                 // S = 512
  const float* trow = traj + (size_t)mrow * (2 * PH);
  const float* drow = dest + b * PH;

  // ---- phase 1: q_proj = [traj | dest] @ Wq^T   (K=768, N=256) ----
  floatx8 acc[NTILES_K];
#pragma unroll
  for (int i = 0; i < NTILES_K; ++i) acc[i] = zero8();
  for (int kb = 0; kb < 3 * PH; kb += 32) {
    const float* ar = (kb < 2 * PH) ? (trow + kb) : (drow + (kb - 2 * PH));
    bf16x16 af = load_a_frag_f32(ar, half);
#pragma unroll
    for (int nt = 0; nt < NTILES_K; ++nt) {
      bf16x16 bf_ = load_b_frag_f32(Wq + (size_t)(nt * 16 + ln) * (3 * PH) + kb + half * 16);
      acc[nt] = WMMA_BF16(af, bf_, acc[nt]);
    }
  }
#pragma unroll
  for (int nt = 0; nt < NTILES_K; ++nt)
#pragma unroll
    for (int v = 0; v < 8; ++v)
      qproj[(size_t)(m0 + v + 8 * half) * PH + nt * 16 + ln] = acc[nt][v];

  // ---- phase 2: qr = q_proj @ Wr1[:, :256]^T    (K=256, N=64) ----
  floatx8 acc2[NTILES_T];
#pragma unroll
  for (int i = 0; i < NTILES_T; ++i) acc2[i] = zero8();
  const float* qrow_in = qproj + (size_t)mrow * PH;
  for (int kb = 0; kb < PH; kb += 32) {
    bf16x16 af = load_a_frag_f32(qrow_in + kb, half);
#pragma unroll
    for (int nt = 0; nt < NTILES_T; ++nt) {
      bf16x16 bf_ = load_b_frag_f32(Wr1 + (size_t)(nt * 16 + ln) * (PH + 1) + kb + half * 16);
      acc2[nt] = WMMA_BF16(af, bf_, acc2[nt]);
    }
  }
#pragma unroll
  for (int nt = 0; nt < NTILES_T; ++nt)
#pragma unroll
    for (int v = 0; v < 8; ++v)
      qrw[(size_t)(m0 + v + 8 * half) * 64 + nt * 16 + ln] = acc2[nt][v];

  // ---- phase 3: t1 = traj @ Wt1^T (K=512, N=64); t1g = sum gelu(t1)*Wt3a ----
  floatx8 acc3[NTILES_T];
#pragma unroll
  for (int i = 0; i < NTILES_T; ++i) acc3[i] = zero8();
  for (int kb = 0; kb < 2 * PH; kb += 32) {
    bf16x16 af = load_a_frag_f32(trow + kb, half);
#pragma unroll
    for (int nt = 0; nt < NTILES_T; ++nt) {
      bf16x16 bf_ = load_b_frag_f32(Wt1 + (size_t)(nt * 16 + ln) * (2 * PH) + kb + half * 16);
      acc3[nt] = WMMA_BF16(af, bf_, acc3[nt]);
    }
  }
  float p[8];
#pragma unroll
  for (int v = 0; v < 8; ++v) p[v] = 0.f;
#pragma unroll
  for (int nt = 0; nt < NTILES_T; ++nt) {
    int n = nt * 16 + ln;
    float bo = bt1[n], w3 = Wt3[n];
#pragma unroll
    for (int v = 0; v < 8; ++v) p[v] += geluf(acc3[nt][v] + bo) * w3;
  }
#pragma unroll
  for (int v = 0; v < 8; ++v) {
    float s = p[v];
    for (int off = 8; off; off >>= 1) s += __shfl_xor(s, off, 16);
    if (ln == 0) t1g[m0 + v + 8 * half] = s;
  }
}

// ---- main kernel: fused cand GEMMs + full epilogue -------------------------
__global__ __launch_bounds__(256) void nav_main(
    const float* __restrict__ cand, const float* __restrict__ dist,
    const float* __restrict__ ang,  const __bf16* __restrict__ Wbf,
    const float* __restrict__ Wv,
    const float* __restrict__ bt2,  const float* __restrict__ Wt3,
    const float* __restrict__ bt3,  const float* __restrict__ Wr1,
    const float* __restrict__ br1,  const float* __restrict__ Wr2,
    const float* __restrict__ br2,  const float* __restrict__ vecd,
    const float* __restrict__ veca, const float* __restrict__ kcst,
    const float* __restrict__ qproj, const float* __restrict__ qrw,
    const float* __restrict__ t1g,
    float* __restrict__ out_logits, float* __restrict__ out_time,
    float* __restrict__ out_rate) {
  __shared__ __attribute__((aligned(16))) __bf16 ldsW[320 * LDSW_LD]; // 50 KB
  __shared__ float ldsLog[MT];                                        // 512 B

  const int wave = threadIdx.x >> 5;
  const int lane = threadIdx.x & 31;
  const int half = lane >> 4;
  const int ln   = lane & 15;
  const int numTiles = PM / MT;   // 1024

  for (int tile = blockIdx.x; tile < numTiles; tile += gridDim.x) {
    const int m0  = tile * MT;
    const int m0w = m0 + wave * 16;
    const int bs  = m0w >> 5;     // C = 32, 16-row tile stays in one (b,s)

    floatx8 acc[NTILES_ALL];
#pragma unroll
    for (int i = 0; i < NTILES_ALL; ++i) acc[i] = zero8();

    for (int panel = 0; panel < PH / PANEL_K; ++panel) {
      const int kb = panel * PANEL_K;
      __syncthreads();             // previous panel fully consumed
#if HAS_TDM
      // TDM: copy 320 rows x 128 B (32 dwords) tile of Wbf into LDS,
      // hardware-padding each row by 8 dwords to the 160 B LDS pitch.
      if (threadIdx.x == 0) {
        unsigned lds_base =
            (unsigned)(size_t)(__attribute__((address_space(3))) char*)(&ldsW[0]);
        unsigned long long ga = (unsigned long long)(const void*)(Wbf + kb);
        u32x4 g0;
        g0[0] = 1u;                                   // count = 1 (valid D#)
        g0[1] = lds_base;                             // lds_addr
        g0[2] = (unsigned)(ga & 0xFFFFFFFFu);         // global_addr[31:0]
        g0[3] = (unsigned)((ga >> 32) & 0x01FFFFFFu)  // global_addr[56:32]
                | 0x80000000u;                        // type = 2 ("image")
        i32x8 g1;
        g1[0] = (2 << 16)     // data_size = 4 B
              | (1 << 20)     // pad_enable
              | (4 << 22)     // pad_interval: 32 dwords (one 128 B row)
              | (7 << 25);    // pad_amount: 8 dwords (32 B)
        g1[1] = (int)(32u << 16);    // tensor_dim0 = 32 units (lo16)
        g1[2] = (int)(320u << 16);   // tensor_dim0 hi | tensor_dim1 = 320
        g1[3] = (int)(32u << 16);    // tensor_dim1 hi | tile_dim0 = 32
        g1[4] = 320;                 // tile_dim1 = 320, tile_dim2 = 0
        g1[5] = 128;                 // tensor_dim0_stride = 128 units (512 B row)
        g1[6] = 0;
        g1[7] = 0;
        i32x4 gz4; gz4[0] = gz4[1] = gz4[2] = gz4[3] = 0;
        i32x8 gz8;
#pragma unroll
        for (int i = 0; i < 8; ++i) gz8[i] = 0;
        __builtin_amdgcn_tensor_load_to_lds(g0, g1, gz4, gz4, gz8, 0);
        __builtin_amdgcn_s_wait_tensorcnt(0);
      }
#else
      for (int idx = threadIdx.x; idx < 320 * PANEL_K; idx += 256) {
        int n = idx >> 6, kk = idx & 63;
        ldsW[n * LDSW_LD + kk] = Wbf[(size_t)n * 256 + kb + kk];
      }
#endif
      __syncthreads();
#pragma unroll
      for (int ks = 0; ks < 2; ++ks) {
        const float* arow = cand + (size_t)(m0w + ln) * PH + kb + ks * 32;
        bf16x16 af = load_a_frag_f32(arow, half);
#pragma unroll
        for (int nt = 0; nt < NTILES_ALL; ++nt) {
          bf16x16 bf_ = load_b_frag_lds(&ldsW[(nt * 16 + ln) * LDSW_LD + ks * 32 + half * 16]);
          acc[nt] = WMMA_BF16(af, bf_, acc[nt]);
        }
      }
    }

    // ---- fused epilogue (register-resident) ----
    float d8[8], a8[8];
#pragma unroll
    for (int v = 0; v < 8; ++v) {
      int m = m0w + v + 8 * half;
      d8[v] = dist[m]; a8[v] = ang[m];
    }
    float psum[8], tsum[8];
#pragma unroll
    for (int v = 0; v < 8; ++v) { psum[v] = 0.f; tsum[v] = 0.f; }

#pragma unroll
    for (int nt = 0; nt < NTILES_K; ++nt) {
      int n = nt * 16 + ln;
      float vd = vecd[n], va = veca[n];
      float kq = kcst[n] + qproj[(size_t)bs * PH + n];
      float wv = Wv[n];
#pragma unroll
      for (int v = 0; v < 8; ++v) {
        float x = acc[nt][v] + d8[v] * vd + a8[v] * va + kq;
        psum[v] += tanhf(x) * wv;
      }
    }
#pragma unroll
    for (int nt = 0; nt < NTILES_T; ++nt) {
      int o = nt * 16 + ln;
      float bo = bt2[o], w3 = Wt3[64 + o];
#pragma unroll
      for (int v = 0; v < 8; ++v)
        tsum[v] += geluf(acc[NTILES_K + nt][v] + bo) * w3;
    }
    float tg = t1g[bs] + bt3[0];
#pragma unroll
    for (int v = 0; v < 8; ++v) {
      float l = psum[v], t = tsum[v];
      for (int off = 8; off; off >>= 1) {
        l += __shfl_xor(l, off, 16);
        t += __shfl_xor(t, off, 16);
      }
      if (ln == 0) {
        int m = m0w + v + 8 * half;
        out_logits[m] = l;
        out_time[m] = tg + t;
        ldsLog[wave * 16 + v + 8 * half] = l;
      }
    }
    __syncthreads();

    // ---- rate branch: one thread per row of the 128-row tile ----
    if (threadIdx.x < MT) {
      int m = m0 + threadIdx.x;
      int bsr = m >> 5;
      float l = ldsLog[threadIdx.x];
      const float* qrow = qrw + (size_t)bsr * 64;
      float accr = br2[0];
      for (int o = 0; o < 64; ++o) {
        float r1 = geluf(qrow[o] + l * Wr1[(size_t)o * (PH + 1) + PH] + br1[o]);
        accr += r1 * Wr2[o];
      }
      out_rate[m] = sigmoidf_(accr);
    }
  }
}

extern "C" void kernel_launch(void* const* d_in, const int* in_sizes, int n_in,
                              void* d_out, int out_size, void* d_ws, size_t ws_size,
                              hipStream_t stream) {
  (void)in_sizes; (void)n_in; (void)out_size; (void)ws_size;
  const float* traj = (const float*)d_in[0];
  const float* dest = (const float*)d_in[1];
  const float* cand = (const float*)d_in[2];
  const float* dist = (const float*)d_in[3];
  const float* ang  = (const float*)d_in[4];
  const float* Wd   = (const float*)d_in[5];
  const float* bd   = (const float*)d_in[6];
  const float* Wa   = (const float*)d_in[7];
  const float* ba   = (const float*)d_in[8];
  const float* Wq   = (const float*)d_in[9];
  const float* Wk   = (const float*)d_in[10];
  const float* Wv   = (const float*)d_in[11];
  const float* Wt1  = (const float*)d_in[12];
  const float* bt1  = (const float*)d_in[13];
  const float* Wt2  = (const float*)d_in[14];
  const float* bt2  = (const float*)d_in[15];
  const float* Wt3  = (const float*)d_in[16];
  const float* bt3  = (const float*)d_in[17];
  const float* Wr1  = (const float*)d_in[18];
  const float* br1  = (const float*)d_in[19];
  const float* Wr2  = (const float*)d_in[20];
  const float* br2  = (const float*)d_in[21];

  float* ws    = (float*)d_ws;
  float* vecd  = ws;                         // 256
  float* veca  = ws + 256;                   // 256
  float* kcst  = ws + 512;                   // 256
  float* qproj = ws + 768;                   // 4096*256
  float* qrw   = qproj + (size_t)PBS * PH;   // 4096*64
  float* t1g   = qrw + (size_t)PBS * 64;     // 4096
  __bf16* Wbf  = (__bf16*)(t1g + PBS);       // 320*256 bf16

  float* out_logits = (float*)d_out;
  float* out_time   = out_logits + PM;
  float* out_rate   = out_time + PM;

  nav_pre<<<1, 256, 0, stream>>>(Wk, Wd, bd, Wa, ba, vecd, veca, kcst);
  nav_cvtw<<<80, 256, 0, stream>>>(Wk, Wt2, Wbf);
  nav_bs<<<32, 256, 0, stream>>>(traj, dest, Wq, Wr1, Wt1, bt1, Wt3, qproj, qrw, t1g);
  nav_main<<<256, 256, 0, stream>>>(cand, dist, ang, Wbf, Wv, bt2, Wt3, bt3,
                                    Wr1, br1, Wr2, br2, vecd, veca, kcst,
                                    qproj, qrw, t1g, out_logits, out_time, out_rate);
}
